// LSTMStockPredictor_14061722928026
// MI455X (gfx1250) — matrix operand order repaired and checked by reading, not compiled, not run on hardware
//
#include <hip/hip_runtime.h>

typedef __attribute__((ext_vector_type(16))) _Float16 v16h;
typedef __attribute__((ext_vector_type(8)))  float    v8f;

#define SEQ_LEN 131072
#define HID     100
#define NTHREAD 800          // 25 waves * 32 lanes; wave w owns j = 4w..4w+3 for all 4 gates
#define KCH     4            // K padded 100 -> 128, four 32-wide chunks

__launch_bounds__(NTHREAD, 1)
__global__ void lstm_scan_kernel(const float* __restrict__ input_seq,
                                 const float* __restrict__ W_ih,
                                 const float* __restrict__ W_hh,
                                 const float* __restrict__ b_ih,
                                 const float* __restrict__ b_hh,
                                 const float* __restrict__ W_out,
                                 const float* __restrict__ b_out,
                                 float* __restrict__ out)
{
    // One array holds: [0..127] h buf0 (f16), [128..255] h buf1, [256..511] zeros.
    // The zero region is big enough that non-contributing lanes can take the same
    // "+128*(t&1) + 32*cc" address math as lanes 0/16 and still read zeros ->
    // completely branchless, loop-invariant B addressing.
    __shared__ __align__(32) _Float16 shh[512];
    __shared__ float h32[128];

    const int  tid  = threadIdx.x;
    const int  lane = tid & 31;
    const int  w    = tid >> 5;            // wave id 0..24
    const bool hi   = (lane & 16) != 0;    // upper half-wave
    const int  tl   = lane & 15;

    // ---- init LDS ----
    for (int i = tid; i < 512; i += NTHREAD) shh[i] = (_Float16)0.0f;
    for (int i = tid; i < 128; i += NTHREAD) h32[i] = 0.0f;

    // ---- preload this wave's A tiles (permuted W_hh rows, f16, K zero-padded) ----
    // tile row m (0..15): gate = m/4, j = 4w + m%4  ->  global row = (m/4)*HID + 4w + (m%4)
    // A 16x32 f16, contiguous k order chosen here (B uses the same): lanes 0-15 row=lane, halves 0..15 = K 0..15;
    //                     lanes 16-31 row=lane-16, halves 0..15 = K 16..31.
    v16h A[KCH];
    {
        const int grow = (tl >> 2) * HID + 4 * w + (tl & 3);
        for (int cc = 0; cc < KCH; ++cc) {
            v16h a;
#pragma unroll
            for (int hh = 0; hh < 16; ++hh) {
                int kl = hh + (hi ? 16 : 0);
                int k  = cc * 32 + kl;
                float v = (k < HID) ? W_hh[grow * HID + k] : 0.0f;
                a[hh] = (_Float16)v;
            }
            A[cc] = a;
        }
    }

    // ---- per-lane input-gate constants (meaningful on lanes 0 and 16) ----
    float wih[8], bsum[8];
    {
        const int base = hi ? 8 : 0;
#pragma unroll
        for (int r = 0; r < 8; ++r) {
            int tr   = base + r;
            int grow = (tr >> 2) * HID + 4 * w + (tr & 3);
            wih[r]  = W_ih[grow];
            bsum[r] = b_ih[grow] + b_hh[grow];
        }
    }

    // B 32x16 f16: column 0 = h chunk. Lane0 halves = K 0..15 (idx +0),
    // lane16 halves = K 16..31 (idx +16); all other lanes hit the zero region.
    // Per-lane constant base index (halves) -- no selects in the loop.
    const int contrib = (tl == 0);                       // lanes 0 and 16
    const int bbase   = contrib ? ((lane >> 4) << 4) : 256;

    float cst[4]  = {0.f, 0.f, 0.f, 0.f};   // cell state for j=4w..4w+3 (lane 0)
    float hval[4] = {0.f, 0.f, 0.f, 0.f};   // last h (lane 0), published after loop

    __syncthreads();

    for (int t = 0; t < SEQ_LEN; ++t) {
        const int boff = bbase + ((t & 1) << 7);         // ping-pong toggle, uniform add

        __builtin_prefetch(&input_seq[t + 256], 0, 1);   // warm the scalar input stream

        // Two accumulator chains -> half the WMMA latency on the critical path.
        v8f acc0 = {}, acc1 = {};
        {
            v16h b0 = *(const v16h*)&shh[boff +  0];
            v16h b1 = *(const v16h*)&shh[boff + 32];
            v16h b2 = *(const v16h*)&shh[boff + 64];
            v16h b3 = *(const v16h*)&shh[boff + 96];
            acc0 = __builtin_amdgcn_wmma_f32_16x16x32_f16(false, A[0], false, b0,
                                                          (short)0, acc0, false, false);
            acc1 = __builtin_amdgcn_wmma_f32_16x16x32_f16(false, A[1], false, b1,
                                                          (short)0, acc1, false, false);
            acc0 = __builtin_amdgcn_wmma_f32_16x16x32_f16(false, A[2], false, b2,
                                                          (short)0, acc0, false, false);
            acc1 = __builtin_amdgcn_wmma_f32_16x16x32_f16(false, A[3], false, b3,
                                                          (short)0, acc1, false, false);
        }

        const float xt = input_seq[t];

        // D column 0: lane0 VGPRs 0..7 = tile rows 0..7 (i then f),
        //             lane16 VGPRs 0..7 = tile rows 8..15 (g then o).
        float act[8];
#pragma unroll
        for (int r = 0; r < 8; ++r) {
            float v   = (acc0[r] + acc1[r]) + xt * wih[r] + bsum[r];
            float sig = __builtin_amdgcn_rcpf(1.0f + __expf(-v));
            if (r < 4) {
                // only rows 0..3 of the hi half (g gate) need tanh
                float th = 1.0f - 2.0f * __builtin_amdgcn_rcpf(__expf(2.0f * v) + 1.0f);
                act[r] = hi ? th : sig;
            } else {
                act[r] = sig;
            }
        }

        // combine gates intra-wave: lane0 has i,f ; lane16 has g,o (single shfl each)
#pragma unroll
        for (int q = 0; q < 4; ++q) {
            float gq = __shfl(act[q],     16, 32);
            float oq = __shfl(act[4 + q], 16, 32);
            float iq = act[q];
            float fq = act[4 + q];
            float cn = fq * cst[q] + iq * gq;
            cst[q] = cn;
            float tc = 1.0f - 2.0f * __builtin_amdgcn_rcpf(__expf(2.0f * cn) + 1.0f);
            hval[q] = oq * tc;
        }

        // lane0 publishes f16 h for the next step into the other ping-pong buffer
        if (lane == 0) {
            _Float16* hn = &shh[(((t + 1) & 1) << 7) + 4 * w];
#pragma unroll
            for (int q = 0; q < 4; ++q) hn[q] = (_Float16)hval[q];
        }
        __syncthreads();   // single barrier per step
    }

    // publish the final f32 h once, then reduce the output linear layer
    if (lane == 0) {
#pragma unroll
        for (int q = 0; q < 4; ++q) h32[4 * w + q] = hval[q];
    }
    __syncthreads();

    if (tid == 0) {
        float s = b_out[0];
        for (int j = 0; j < HID; ++j) s += W_out[j] * h32[j];
        out[0] = s;
    }
}

extern "C" void kernel_launch(void* const* d_in, const int* in_sizes, int n_in,
                              void* d_out, int out_size, void* d_ws, size_t ws_size,
                              hipStream_t stream) {
    (void)in_sizes; (void)n_in; (void)d_ws; (void)ws_size; (void)out_size;
    const float* input_seq = (const float*)d_in[0];
    const float* W_ih      = (const float*)d_in[1];
    const float* W_hh      = (const float*)d_in[2];
    const float* b_ih      = (const float*)d_in[3];
    const float* b_hh      = (const float*)d_in[4];
    const float* W_out     = (const float*)d_in[5];
    const float* b_out     = (const float*)d_in[6];
    float* out = (float*)d_out;

    // One persistent workgroup owns the whole scan (serial recurrence).
    lstm_scan_kernel<<<1, NTHREAD, 0, stream>>>(input_seq, W_ih, W_hh,
                                                b_ih, b_hh, W_out, b_out, out);
}
